// SP_Attn_17076789969511
// MI455X (gfx1250) — matrix operand order repaired and checked by reading, not compiled, not run on hardware
//
#include <hip/hip_runtime.h>

// Problem constants (B=1, S=2304, C=1280, H=20, d=64)
#define S_LEN 2304
#define C_DIM 1280
#define NH    20
#define DH    64

typedef __attribute__((ext_vector_type(16))) __bf16       v16bf;
typedef __attribute__((ext_vector_type(8)))  float        v8f;
typedef __attribute__((ext_vector_type(4)))  unsigned int u32x4;

union FragU {
    struct { u32x4 a, b; } s;   // 32 bytes = 16 bf16
    v16bf v;
};

__device__ __forceinline__ unsigned short f2bf(float f) {
    unsigned u = __builtin_bit_cast(unsigned, f);
    u += 0x7fffu + ((u >> 16) & 1u);            // round-to-nearest-even
    return (unsigned short)(u >> 16);
}

__device__ __forceinline__ v8f wmma_bf16(v16bf a, v16bf b, v8f c) {
    // v_wmma_f32_16x16x32_bf16 : D = A(16x32) * B(32x16) + C(16x16 f32)
    return __builtin_amdgcn_wmma_f32_16x16x32_bf16(false, a, false, b,
                                                   (short)0, c, false, false);
}

// A-fragment: lane holds row (lm), 16 contiguous K starting at hi*16
__device__ __forceinline__ void loadA(FragU& f, const unsigned short* row,
                                      int k0, int hi) {
    const u32x4* p = (const u32x4*)(row + k0);
    f.s.a = p[2 * hi]; f.s.b = p[2 * hi + 1];
}
// B-fragment: lane holds col (lm), 16 contiguous K starting at k0 (+hi*16 baked
// into `row` by the caller)
__device__ __forceinline__ void loadB(FragU& f, const unsigned short* row, int k0) {
    const u32x4* p = (const u32x4*)(row + k0);
    f.s.a = p[0]; f.s.b = p[1];
}

// ---------------------------------------------------------------- fp32 -> bf16
__global__ void k_cvt_bf16(const float* __restrict__ src,
                           unsigned short* __restrict__ dst, int n) {
    for (int i = blockIdx.x * blockDim.x + threadIdx.x; i < n;
         i += gridDim.x * blockDim.x)
        dst[i] = f2bf(src[i]);
}

// -------------------------------------------- fp32 [K][N] -> bf16 [N][K] (W^T)
__global__ void k_tr_bf16(const float* __restrict__ src,
                          unsigned short* __restrict__ dst, int n) {
    __shared__ float tile[32][33];
    int bx = blockIdx.x * 32, by = blockIdx.y * 32;
    int tx = threadIdx.x, ty = threadIdx.y;
#pragma unroll
    for (int i = ty; i < 32; i += 8)
        tile[i][tx] = src[(size_t)(by + i) * n + bx + tx];
    __syncthreads();
#pragma unroll
    for (int i = ty; i < 32; i += 8)
        dst[(size_t)(bx + i) * n + by + tx] = f2bf(tile[tx][i]);
}

// ------------------------------------------------------------------- GEMM
// C[M,N] = A[M,K](bf16) * Bt[N,K](bf16)^T ; wave tile 32x64, block 256x64.
// K must be a multiple of 64 (ping-pong unroll).
// mode 0: outb[M,N] = bf16(acc*scale)   (Q with 1/sqrt(d), K)
// mode 1: outb[N,M] = bf16(acc)         (V stored transposed)
// mode 2: outf[M,N] = acc + bias[N]     (final projection, fp32)
__global__ __launch_bounds__(256) void k_gemm(
        const unsigned short* __restrict__ A,
        const unsigned short* __restrict__ Bt,
        unsigned short* __restrict__ outb,
        float* __restrict__ outf,
        const float* __restrict__ bias,
        float scale, int M, int N, int K, int mode) {
    int lane = threadIdx.x & 31, w = threadIdx.x >> 5;
    int hi = lane >> 4, lm = lane & 15;
    int m0 = blockIdx.y * 256 + w * 32;     // 8 waves -> 256 rows / block
    int n0 = blockIdx.x * 64;               // 4 N-tiles / wave

    const unsigned short* Arow0 = A + (size_t)(m0 + lm) * K;
    const unsigned short* Arow1 = A + (size_t)(m0 + 16 + lm) * K;
    const unsigned short* Brow[4];
#pragma unroll
    for (int t = 0; t < 4; ++t)
        Brow[t] = Bt + (size_t)(n0 + 16 * t + lm) * K + hi * 16;

    v8f acc[8] = {};
    FragU a0[2], b0[4], a1[2], b1[4];      // ping-pong buffers (no copies)

    auto ldStep = [&](FragU (&fa)[2], FragU (&fb)[4], int k0) {
        loadA(fa[0], Arow0, k0, hi);
        loadA(fa[1], Arow1, k0, hi);
#pragma unroll
        for (int t = 0; t < 4; ++t) loadB(fb[t], Brow[t], k0);
    };
    auto mmStep = [&](FragU (&fa)[2], FragU (&fb)[4]) {
#pragma unroll
        for (int t = 0; t < 4; ++t) {
            acc[t]     = wmma_bf16(fa[0].v, fb[t].v, acc[t]);
            acc[4 + t] = wmma_bf16(fa[1].v, fb[t].v, acc[4 + t]);
        }
    };

    ldStep(a0, b0, 0);
    for (int k0 = 0; k0 < K; k0 += 64) {
        ldStep(a1, b1, k0 + 32);           // other buffer's loads in flight
        mmStep(a0, b0);
        if (k0 + 64 < K) ldStep(a0, b0, k0 + 64);
        mmStep(a1, b1);
    }

#pragma unroll
    for (int mf = 0; mf < 2; ++mf)
#pragma unroll
        for (int t = 0; t < 4; ++t)
#pragma unroll
            for (int r = 0; r < 8; ++r) {
                int row = m0 + 16 * mf + hi * 8 + r;
                int col = n0 + 16 * t + lm;
                float vv = acc[4 * mf + t][r];
                if (mode == 0)      outb[(size_t)row * N + col] = f2bf(vv * scale);
                else if (mode == 1) outb[(size_t)col * M + row] = f2bf(vv);
                else                outf[(size_t)row * N + col] = vv + bias[col];
            }
}

// --------------------------------------------------------- flash attention
// Qb[S,C] (pre-scaled bf16), Kb[S,C] bf16, Vt[C,S] bf16, mask[S,S] int,
// Ab[S,C] bf16 out.  Block = 8 waves, each wave owns 16 queries of head h.
// S_LEN/32 is even -> exact ping-pong unroll over 64-key super-chunks.
__global__ __launch_bounds__(256) void k_attn(
        const unsigned short* __restrict__ Qb,
        const unsigned short* __restrict__ Kb,
        const unsigned short* __restrict__ Vt,
        const int* __restrict__ mask,
        unsigned short* __restrict__ Ab) {
    __shared__ __align__(16) unsigned short lds[8][16][32];  // per-wave P tile

    int lane = threadIdx.x & 31, w = threadIdx.x >> 5;
    int hi = lane >> 4, lm = lane & 15;
    int q0 = blockIdx.x * 128 + w * 16;
    int ch = blockIdx.y * DH;                       // head column base

    // Q fragments for d=0..31 and d=32..63 (loaded once)
    FragU fq[2];
#pragma unroll
    for (int j = 0; j < 2; ++j)
        loadA(fq[j], Qb + (size_t)(q0 + lm) * C_DIM + ch + 32 * j, 0, hi);

    float mprev[8], lsum[8];
#pragma unroll
    for (int r = 0; r < 8; ++r) { mprev[r] = -1e30f; lsum[r] = 0.0f; }
    v8f o[4] = {};

    auto loadK = [&](FragU (&fk)[4], int kk) {
#pragma unroll
        for (int st = 0; st < 2; ++st)
#pragma unroll
            for (int dj = 0; dj < 2; ++dj)
                loadB(fk[st * 2 + dj],
                      Kb + (size_t)(kk + 16 * st + lm) * C_DIM
                          + ch + 32 * dj + hi * 16, 0);
    };

    // one 32-key chunk: uses `cur`, loads `nxt` for kk+32 (if ldn)
    auto step = [&](int kk, FragU (&cur)[4], FragU (&nxt)[4], bool ldn) {
        // mask words first so they overlap the score WMMAs
        int mk[2][8];
#pragma unroll
        for (int st = 0; st < 2; ++st)
#pragma unroll
            for (int r = 0; r < 8; ++r)
                mk[st][r] = mask[(size_t)(q0 + hi * 8 + r) * S_LEN
                                 + kk + 16 * st + lm];
        if (ldn) {
            loadK(nxt, kk + 32);
            __builtin_prefetch(Vt + (size_t)(ch + lm) * S_LEN + kk + 32, 0, 1);
        }

        // scores: 2 key tiles x 2 k-steps over d
        v8f sc[2] = {};
#pragma unroll
        for (int st = 0; st < 2; ++st)
#pragma unroll
            for (int dj = 0; dj < 2; ++dj)
                sc[st] = wmma_bf16(fq[dj].v, cur[st * 2 + dj].v, sc[st]);

        // additive mask (finite -inf surrogate; diagonal keeps rows alive)
#pragma unroll
        for (int st = 0; st < 2; ++st)
#pragma unroll
            for (int r = 0; r < 8; ++r)
                if (mk[st][r] == 0) sc[st][r] = -1e30f;

        // online softmax: per-row max/sum across the 16 lanes of each half
#pragma unroll
        for (int r = 0; r < 8; ++r) {
            float cm = fmaxf(sc[0][r], sc[1][r]);
#pragma unroll
            for (int off = 1; off < 16; off <<= 1)
                cm = fmaxf(cm, __shfl_xor(cm, off, 32));
            float mn = fmaxf(mprev[r], cm);
            float alpha = __expf(mprev[r] - mn);
            float p0 = __expf(sc[0][r] - mn);
            float p1 = __expf(sc[1][r] - mn);
            sc[0][r] = p0; sc[1][r] = p1;
            float ls = p0 + p1;
#pragma unroll
            for (int off = 1; off < 16; off <<= 1)
                ls += __shfl_xor(ls, off, 32);
            lsum[r] = lsum[r] * alpha + ls;
            mprev[r] = mn;
#pragma unroll
            for (int t = 0; t < 4; ++t) o[t][r] *= alpha;
        }

        // V fragments issued before the P LDS round-trip (overlap DS ops)
        FragU fv[4];
#pragma unroll
        for (int t = 0; t < 4; ++t)
            loadB(fv[t], Vt + (size_t)(ch + 16 * t + lm) * S_LEN + kk + hi * 16, 0);

        // re-stripe P (C/D layout) -> A layout via per-wave LDS tile
#pragma unroll
        for (int st = 0; st < 2; ++st)
#pragma unroll
            for (int r = 0; r < 8; ++r)
                lds[w][hi * 8 + r][st * 16 + lm] = f2bf(sc[st][r]);
        FragU fp;
        {
            const u32x4* pp = (const u32x4*)(&lds[w][lm][0]);
            fp.s.a = pp[2 * hi]; fp.s.b = pp[2 * hi + 1];   // same-wave DS: in order
        }

        // O += P(16x32) * V(32x64)
#pragma unroll
        for (int t = 0; t < 4; ++t)
            o[t] = wmma_bf16(fp.v, fv[t].v, o[t]);
    };

    FragU fk0[4], fk1[4];                  // ping-pong K fragments
    loadK(fk0, 0);
    for (int kk = 0; kk < S_LEN; kk += 64) {
        step(kk,      fk0, fk1, true);               // loads fk1 for kk+32
        step(kk + 32, fk1, fk0, kk + 64 < S_LEN);    // loads fk0 for kk+64
    }

    // normalize and store attention output (bf16, [S,C] layout)
#pragma unroll
    for (int r = 0; r < 8; ++r) {
        float inv = 1.0f / lsum[r];
        int row = q0 + hi * 8 + r;
#pragma unroll
        for (int t = 0; t < 4; ++t)
            Ab[(size_t)row * C_DIM + ch + 16 * t + lm] = f2bf(o[t][r] * inv);
    }
}

// ------------------------------------------------------------------ driver
extern "C" void kernel_launch(void* const* d_in, const int* in_sizes, int n_in,
                              void* d_out, int out_size, void* d_ws, size_t ws_size,
                              hipStream_t stream) {
    const float* X    = (const float*)d_in[0];   // [S,C]
    const int*   mask = (const int*)  d_in[1];   // [S,S]
    const float* Wq   = (const float*)d_in[2];
    const float* Wk   = (const float*)d_in[3];
    const float* Wv   = (const float*)d_in[4];
    const float* Wo   = (const float*)d_in[5];
    const float* bo   = (const float*)d_in[6];
    float* out        = (float*)d_out;

    char* ws = (char*)d_ws;
    const size_t szXC = (size_t)S_LEN * C_DIM * 2;   // bf16 [S,C]
    const size_t szW  = (size_t)C_DIM * C_DIM * 2;   // bf16 [C,C]
    unsigned short* Xb  = (unsigned short*)(ws);                 // reused as Ab
    unsigned short* WqT = (unsigned short*)(ws + szXC);
    unsigned short* WkT = (unsigned short*)(ws + szXC + szW);
    unsigned short* WvT = (unsigned short*)(ws + szXC + 2 * szW);
    unsigned short* WoT = (unsigned short*)(ws + szXC + 3 * szW);
    unsigned short* Qb  = (unsigned short*)(ws + szXC + 4 * szW);
    unsigned short* Kb  = (unsigned short*)(ws + szXC + 4 * szW + szXC);
    unsigned short* Vt  = (unsigned short*)(ws + szXC + 4 * szW + 2 * szXC);
    unsigned short* Ab  = Xb;   // Xb dead after QKV projections

    // 1) convert activations + transpose-convert weights to bf16
    int nX = S_LEN * C_DIM;
    k_cvt_bf16<<<(nX + 255) / 256, 256, 0, stream>>>(X, Xb, nX);
    dim3 trb(32, 8), trg(C_DIM / 32, C_DIM / 32);
    k_tr_bf16<<<trg, trb, 0, stream>>>(Wq, WqT, C_DIM);
    k_tr_bf16<<<trg, trb, 0, stream>>>(Wk, WkT, C_DIM);
    k_tr_bf16<<<trg, trb, 0, stream>>>(Wv, WvT, C_DIM);
    k_tr_bf16<<<trg, trb, 0, stream>>>(Wo, WoT, C_DIM);

    // 2) QKV projections (Q pre-scaled by 1/sqrt(d); V stored transposed)
    dim3 gg(C_DIM / 64, S_LEN / 256);
    k_gemm<<<gg, 256, 0, stream>>>(Xb, WqT, Qb, nullptr, nullptr,
                                   0.125f, S_LEN, C_DIM, C_DIM, 0);
    k_gemm<<<gg, 256, 0, stream>>>(Xb, WkT, Kb, nullptr, nullptr,
                                   1.0f, S_LEN, C_DIM, C_DIM, 0);
    k_gemm<<<gg, 256, 0, stream>>>(Xb, WvT, Vt, nullptr, nullptr,
                                   1.0f, S_LEN, C_DIM, C_DIM, 1);

    // 3) masked flash attention, all heads
    dim3 ag(S_LEN / 128, NH);
    k_attn<<<ag, 256, 0, stream>>>(Qb, Kb, Vt, mask, Ab);

    // 4) output projection + bias, fp32 result
    k_gemm<<<gg, 256, 0, stream>>>(Ab, WoT, nullptr, out, bo,
                                   1.0f, S_LEN, C_DIM, C_DIM, 2);
}
